// DiTBlock_43834436223582
// MI455X (gfx1250) — matrix operand order repaired and checked by reading, not compiled, not run on hardware
//
#include <hip/hip_runtime.h>
#include <hip/hip_bf16.h>
#include <stdint.h>
#include <stddef.h>

// ---------------------------------------------------------------------------
// DiT block forward for MI455X (gfx1250, wave32, WMMA, TDM).
// All GEMMs / attention matmuls: v_wmma_f32_16x16x32_bf16 (bf16 in, fp32 acc).
// LayerNorm / softmax / GELU / residuals in fp32 VALU.
// GEMM A-tiles staged into LDS by the Tensor Data Mover.
// ---------------------------------------------------------------------------

typedef __bf16 bf16_t;
typedef __bf16 v16bf __attribute__((ext_vector_type(16)));
typedef float  v8f   __attribute__((ext_vector_type(8)));
typedef unsigned int v4u __attribute__((ext_vector_type(4)));
typedef int v8i __attribute__((ext_vector_type(8)));
typedef int v4i __attribute__((ext_vector_type(4)));

#define D_    1024
#define L_    2048
#define H_    16
#define DH_   64
#define FF_   2048
#define B_    2
#define ROWS_ 4096            // B_*L_
#define ADA_  6144            // 6*D_

union AFrag { v16bf v; bf16_t h[16]; uint4 q[2]; };

__device__ __forceinline__ v8f wmma_bf16(v16bf a, v16bf b, v8f c) {
  return __builtin_amdgcn_wmma_f32_16x16x32_bf16(false, a, false, b, (short)0, c,
                                                 false, false);
}

// ---- DPP butterfly reductions within each 16-lane row (wave32) -------------
__device__ __forceinline__ float red16_max(float x) {
  x = fmaxf(x, __int_as_float(__builtin_amdgcn_update_dpp(
          0, __float_as_int(x), 0xB1 /*quad_perm(1,0,3,2)*/, 0xF, 0xF, true)));
  x = fmaxf(x, __int_as_float(__builtin_amdgcn_update_dpp(
          0, __float_as_int(x), 0x4E /*quad_perm(2,3,0,1)*/, 0xF, 0xF, true)));
  x = fmaxf(x, __int_as_float(__builtin_amdgcn_update_dpp(
          0, __float_as_int(x), 0x141 /*row_half_mirror*/, 0xF, 0xF, true)));
  x = fmaxf(x, __int_as_float(__builtin_amdgcn_update_dpp(
          0, __float_as_int(x), 0x140 /*row_mirror*/, 0xF, 0xF, true)));
  return x;
}
__device__ __forceinline__ float red16_add(float x) {
  x += __int_as_float(__builtin_amdgcn_update_dpp(
          0, __float_as_int(x), 0xB1, 0xF, 0xF, true));
  x += __int_as_float(__builtin_amdgcn_update_dpp(
          0, __float_as_int(x), 0x4E, 0xF, 0xF, true));
  x += __int_as_float(__builtin_amdgcn_update_dpp(
          0, __float_as_int(x), 0x141, 0xF, 0xF, true));
  x += __int_as_float(__builtin_amdgcn_update_dpp(
          0, __float_as_int(x), 0x140, 0xF, 0xF, true));
  return x;
}

// ---- Tensor Data Mover: 2D tile (element size 2B) global -> LDS ------------
// rowlen/nrows  : tensor dims (elements / rows)
// stride        : elements between consecutive tensor rows
// tile_w/tile_h : tile dims (elements / rows); LDS dest is contiguous
__device__ __forceinline__ void tdm_load_2d_bf16(unsigned lds_off, const void* gptr,
                                                 unsigned rowlen, unsigned nrows,
                                                 unsigned stride,
                                                 unsigned tile_w, unsigned tile_h)
{
  unsigned long long ga = (unsigned long long)(uintptr_t)gptr;
  v4u g0;
  g0[0] = 1u;                                         // count=1, user mode
  g0[1] = lds_off;                                    // lds_addr (bytes)
  g0[2] = (unsigned)(ga & 0xFFFFFFFFu);               // global_addr[31:0]
  g0[3] = (unsigned)((ga >> 32) & 0x01FFFFFFu) | (2u << 30);  // addr hi | type=2
  v8i g1;
  g1[0] = (int)(1u << 16);                            // data_size=1 (2 bytes)
  g1[1] = (int)((rowlen & 0xFFFFu) << 16);            // tensor_dim0[15:0]
  g1[2] = (int)((rowlen >> 16) | ((nrows & 0xFFFFu) << 16));  // dim0 hi | dim1 lo
  g1[3] = (int)((nrows >> 16) | (tile_w << 16));      // dim1 hi | tile_dim0
  g1[4] = (int)tile_h;                                // tile_dim1 (tile_dim2=0)
  g1[5] = (int)stride;                                // tensor_dim0_stride[31:0]
  g1[6] = 0;
  g1[7] = 0;
  v4i z4 = {0, 0, 0, 0};
  v8i z8 = {0, 0, 0, 0, 0, 0, 0, 0};
  __builtin_amdgcn_tensor_load_to_lds(g0, g1, z4, z4, z8, 0);
}

// ---------------------------------------------------------------------------
// adaLN params: ada[b][j] = silu(c[b]) . Wada[:,j] + bada[j]
// ---------------------------------------------------------------------------
__global__ __launch_bounds__(256)
void k_ada(const float* __restrict__ c, const float* __restrict__ Wada,
           const float* __restrict__ bada, float* __restrict__ ada)
{
  int wave = threadIdx.x >> 5, lane = threadIdx.x & 31;
  int task = blockIdx.x * 8 + wave;            // 12288 tasks
  int b = task / ADA_;
  int j = task - b * ADA_;
  float s = 0.f;
  for (int kk = lane; kk < D_; kk += 32) {
    float cv  = c[b * D_ + kk];
    float sil = cv / (1.f + __expf(-cv));      // silu
    s += sil * Wada[(size_t)kk * ADA_ + j];
  }
#pragma unroll
  for (int off = 16; off >= 1; off >>= 1) s += __shfl_xor(s, off, 32);
  if (lane == 0) ada[(size_t)b * ADA_ + j] = s + bada[j];
}

// ---------------------------------------------------------------------------
// fp32 -> bf16 weight cast
// ---------------------------------------------------------------------------
__global__ __launch_bounds__(256)
void k_f2bf(const float* __restrict__ s, bf16_t* __restrict__ d, int n)
{
  int i = blockIdx.x * 256 + threadIdx.x;
  if (i < n) d[i] = (bf16_t)s[i];
}

// ---------------------------------------------------------------------------
// fused LayerNorm (no affine) + adaLN modulate + cast to bf16
// ---------------------------------------------------------------------------
__global__ __launch_bounds__(256)
void k_lnmod(const float* __restrict__ x, const float* __restrict__ ada,
             int shift_off, int scale_off, bf16_t* __restrict__ out)
{
  __shared__ float red[8];
  int row  = blockIdx.x;
  int b    = row >> 11;                      // L_ = 2048
  int wave = threadIdx.x >> 5, lane = threadIdx.x & 31;
  const float* xr = x + (size_t)row * D_;

  float v[4];
  float s = 0.f;
#pragma unroll
  for (int i = 0; i < 4; ++i) { v[i] = xr[threadIdx.x + i * 256]; s += v[i]; }
#pragma unroll
  for (int off = 16; off >= 1; off >>= 1) s += __shfl_xor(s, off, 32);
  if (lane == 0) red[wave] = s;
  __syncthreads();
  float tot = 0.f;
#pragma unroll
  for (int i = 0; i < 8; ++i) tot += red[i];
  float mu = tot * (1.f / (float)D_);
  __syncthreads();

  float s2 = 0.f;
#pragma unroll
  for (int i = 0; i < 4; ++i) { float d = v[i] - mu; s2 += d * d; }
#pragma unroll
  for (int off = 16; off >= 1; off >>= 1) s2 += __shfl_xor(s2, off, 32);
  if (lane == 0) red[wave] = s2;
  __syncthreads();
  float tv = 0.f;
#pragma unroll
  for (int i = 0; i < 8; ++i) tv += red[i];
  float rstd = rsqrtf(tv * (1.f / (float)D_) + 1e-6f);

  const float* ab = ada + (size_t)b * ADA_;
#pragma unroll
  for (int i = 0; i < 4; ++i) {
    int cidx = threadIdx.x + i * 256;
    float sc = 1.f + ab[scale_off + cidx];
    float sh = ab[shift_off + cidx];
    out[(size_t)row * D_ + cidx] = (bf16_t)((v[i] - mu) * rstd * sc + sh);
  }
}

// ---------------------------------------------------------------------------
// WMMA GEMM: C[M,N] = A[M,K](bf16) * Bw[K,N](bf16) + bias, epilogue by MODE.
// Block tile 256x64, 8 waves; wave w -> rows [32w,32w+32), all 64 cols
// (2x4 accumulators). A tile staged by TDM; transposed-B tile staged manually.
//   MODE 0: q/k scatter  out[b,h,l,dh] (bf16)
//   MODE 1: v transposed out[b,h,dh,l] (bf16)
//   MODE 2: outf = xres + gate_msa * C (fp32)
//   MODE 3: outh = gelu_exact(C)       (bf16)
//   MODE 4: outf = xres + gate_mlp * C (fp32)
// ---------------------------------------------------------------------------
#define TM 256
#define TN 64
#define TK 32

template<int MODE>
__global__ __launch_bounds__(256)
void k_gemm(const bf16_t* __restrict__ A, const bf16_t* __restrict__ Bw,
            const float* __restrict__ bias, int M, int N, int K,
            float* __restrict__ outf, bf16_t* __restrict__ outh,
            const float* __restrict__ xres,
            const float* __restrict__ ada, int gate_off)
{
  __shared__ bf16_t As[TM][TK];              // 16 KB, TDM destination
  __shared__ bf16_t Bs[TN][TK + 8];          // transposed: [n][k]

  int tid  = threadIdx.x;
  int wave = tid >> 5, lane = tid & 31;
  int half = lane >> 4;
  int m0   = blockIdx.y * TM;
  int n0   = blockIdx.x * TN;
  unsigned as_off = (unsigned)(uintptr_t)(void*)&As[0][0];

  v8f acc[2][4];
#pragma unroll
  for (int mt = 0; mt < 2; ++mt)
#pragma unroll
    for (int nt = 0; nt < 4; ++nt) acc[mt][nt] = v8f{};

  for (int k0 = 0; k0 < K; k0 += TK) {
    // TDM: A tile 256 rows x 32 cols (bf16) -> contiguous LDS
    if (wave == 0)
      tdm_load_2d_bf16(as_off, A + (size_t)m0 * K + k0,
                       (unsigned)K, (unsigned)M, (unsigned)K, TK, TM);

    // stage B transposed: 32x64 bf16, 8 elems per thread, scatter to Bs[n][k]
    {
      int kr = tid >> 3;
      int nc = (tid & 7) * 8;
      union { uint4 u; bf16_t h[8]; } t;
      t.u = *(const uint4*)(Bw + (size_t)(k0 + kr) * N + n0 + nc);
#pragma unroll
      for (int i = 0; i < 8; ++i) Bs[nc + i][kr] = t.h[i];
    }
    if (wave == 0) __builtin_amdgcn_s_wait_tensorcnt((short)0);
    __syncthreads();

    AFrag a[2];
#pragma unroll
    for (int mt = 0; mt < 2; ++mt) {
      int row = wave * 32 + mt * 16 + (lane & 15);
      int kb  = half * 8;
      a[mt].q[0] = *(const uint4*)&As[row][kb];
      a[mt].q[1] = *(const uint4*)&As[row][kb + 16];
    }
#pragma unroll
    for (int nt = 0; nt < 4; ++nt) {
      AFrag bfb;
      int ncol = nt * 16 + (lane & 15);
      int kb   = half * 8;
      bfb.q[0] = *(const uint4*)&Bs[ncol][kb];
      bfb.q[1] = *(const uint4*)&Bs[ncol][kb + 16];
#pragma unroll
      for (int mt = 0; mt < 2; ++mt)
        acc[mt][nt] = wmma_bf16(a[mt].v, bfb.v, acc[mt][nt]);
    }
    __syncthreads();
  }

  // epilogue  (C layout: vgpr r, lane -> M = r + 8*half, N = lane&15)
#pragma unroll
  for (int mt = 0; mt < 2; ++mt) {
#pragma unroll
    for (int nt = 0; nt < 4; ++nt) {
#pragma unroll
      for (int r = 0; r < 8; ++r) {
        int m = m0 + wave * 32 + mt * 16 + r + half * 8;
        int n = n0 + nt * 16 + (lane & 15);
        float val = acc[mt][nt][r] + bias[n];
        if (MODE == 0) {                       // [B,H,L,DH]
          int b = m >> 11, l = m & (L_ - 1);
          int h = n >> 6,  dh = n & (DH_ - 1);
          outh[(((size_t)b * H_ + h) * L_ + l) * DH_ + dh] = (bf16_t)val;
        } else if (MODE == 1) {                // [B,H,DH,L]
          int b = m >> 11, l = m & (L_ - 1);
          int h = n >> 6,  dh = n & (DH_ - 1);
          outh[(((size_t)b * H_ + h) * DH_ + dh) * L_ + l] = (bf16_t)val;
        } else if (MODE == 2 || MODE == 4) {   // gated residual, fp32
          int b = m >> 11;
          float g = ada[(size_t)b * ADA_ + gate_off + n];
          outf[(size_t)m * N + n] = xres[(size_t)m * N + n] + g * val;
        } else if (MODE == 3) {                // exact GELU -> bf16
          float gl = 0.5f * val * (1.f + erff(val * 0.70710678118654752f));
          outh[(size_t)m * N + n] = (bf16_t)gl;
        }
      }
    }
  }
}

// ---------------------------------------------------------------------------
// attention: one wave per (b,h,16-query tile); flash-style online softmax
// over 32-key blocks. q,k: [B,H,L,DH] bf16; vt: [B,H,DH,L] bf16.
// Row reductions via DPP16 butterflies; P relaid through per-wave bf16 LDS.
// Output applies the "mix" permutation: l' = h*128 + q/16, d' = (q&15)*64+dh.
// ---------------------------------------------------------------------------
__global__ __launch_bounds__(256)
void k_attn(const bf16_t* __restrict__ q, const bf16_t* __restrict__ k,
            const bf16_t* __restrict__ vt, bf16_t* __restrict__ om)
{
  __shared__ bf16_t plds[8][16][32];         // per-wave P scratch (bf16)

  int wave = threadIdx.x >> 5, lane = threadIdx.x & 31;
  int half = lane >> 4;
  int task = blockIdx.x * 8 + wave;          // 4096 tasks
  int b    = task >> 11;                     // H_*128 = 2048 per batch
  int rem  = task & 2047;
  int h    = rem >> 7;
  int qbase = (rem & 127) * 16;

  size_t bh = (size_t)b * H_ + h;
  const bf16_t* qp = q  + bh * ((size_t)L_ * DH_);
  const bf16_t* kp = k  + bh * ((size_t)L_ * DH_);
  const bf16_t* vp = vt + bh * ((size_t)DH_ * L_);

  // Q A-fragments for dh 0..31 (a0) and 32..63 (a1)
  AFrag a0, a1;
  {
    const bf16_t* base = qp + (size_t)(qbase + (lane & 15)) * DH_;
    int kb = half * 8;
    a0.q[0] = *(const uint4*)(base + kb);
    a0.q[1] = *(const uint4*)(base + kb + 16);
    a1.q[0] = *(const uint4*)(base + 32 + kb);
    a1.q[1] = *(const uint4*)(base + 32 + kb + 16);
  }

  float mprev[8], lprev[8];
  v8f o[4] = {v8f{}, v8f{}, v8f{}, v8f{}};
#pragma unroll
  for (int r = 0; r < 8; ++r) { mprev[r] = -3.0e38f; lprev[r] = 0.f; }

  for (int jb = 0; jb < L_; jb += 32) {
    v8f s0 = {}, s1 = {};
    // scores for keys jb..jb+15 (s0) and jb+16..jb+31 (s1)
#pragma unroll
    for (int t = 0; t < 2; ++t) {
      AFrag b0, b1;
      const bf16_t* kr = kp + (size_t)(jb + t * 16 + (lane & 15)) * DH_;
      int kk = half * 8;
      b0.q[0] = *(const uint4*)(kr + kk);
      b0.q[1] = *(const uint4*)(kr + kk + 16);
      b1.q[0] = *(const uint4*)(kr + 32 + kk);
      b1.q[1] = *(const uint4*)(kr + 32 + kk + 16);
      if (t == 0) { s0 = wmma_bf16(a0.v, b0.v, s0); s0 = wmma_bf16(a1.v, b1.v, s0); }
      else        { s1 = wmma_bf16(a0.v, b0.v, s1); s1 = wmma_bf16(a1.v, b1.v, s1); }
    }

    // online softmax (rows: M = r + 8*half; DPP reduction across 16 lanes)
    float alpha[8];
#pragma unroll
    for (int r = 0; r < 8; ++r) {
      float v0 = s0[r] * 0.125f, v1 = s1[r] * 0.125f;   // 1/sqrt(64)
      float t  = red16_max(fmaxf(v0, v1));
      float mn = fmaxf(mprev[r], t);
      alpha[r] = (mprev[r] <= -3.0e38f) ? 0.f : __expf(mprev[r] - mn);
      float p0 = __expf(v0 - mn), p1 = __expf(v1 - mn);
      s0[r] = p0; s1[r] = p1;
      float rs = red16_add(p0 + p1);
      lprev[r] = alpha[r] * lprev[r] + rs;
      mprev[r] = mn;
    }
#pragma unroll
    for (int nt = 0; nt < 4; ++nt)
#pragma unroll
      for (int r = 0; r < 8; ++r) o[nt][r] *= alpha[r];

    // P (16x32) -> A-fragment layout via per-wave bf16 LDS
#pragma unroll
    for (int r = 0; r < 8; ++r) {
      plds[wave][r + half * 8][lane & 15]        = (bf16_t)s0[r];
      plds[wave][r + half * 8][16 + (lane & 15)] = (bf16_t)s1[r];
    }
    asm volatile("s_wait_dscnt 0" ::: "memory");
    AFrag pa;
    {
      int row = lane & 15;
      int kb  = half * 8;
      pa.q[0] = *(const uint4*)&plds[wave][row][kb];
      pa.q[1] = *(const uint4*)&plds[wave][row][kb + 16];
    }

    // AV: o[q, dh] += P[q, key] * V[key, dh] via transposed V
#pragma unroll
    for (int nt = 0; nt < 4; ++nt) {
      AFrag vb;
      const bf16_t* vr = vp + (size_t)(nt * 16 + (lane & 15)) * L_ + jb + half * 8;
      vb.q[0] = *(const uint4*)vr;
      vb.q[1] = *(const uint4*)(vr + 16);
      o[nt] = wmma_bf16(pa.v, vb.v, o[nt]);
    }
  }

  // finalize + mix-permuted store
  float inv[8];
#pragma unroll
  for (int r = 0; r < 8; ++r) inv[r] = 1.f / lprev[r];
#pragma unroll
  for (int nt = 0; nt < 4; ++nt) {
#pragma unroll
    for (int r = 0; r < 8; ++r) {
      float val = o[nt][r] * inv[r];
      int qg = qbase + r + half * 8;
      int dh = nt * 16 + (lane & 15);
      int lp = h * 128 + (qg >> 4);
      int dp = ((qg & 15) << 6) + dh;
      om[((size_t)b * L_ + lp) * D_ + dp] = (bf16_t)val;
    }
  }
}

// ---------------------------------------------------------------------------
// host-side orchestration
// ---------------------------------------------------------------------------
extern "C" void kernel_launch(void* const* d_in, const int* in_sizes, int n_in,
                              void* d_out, int out_size, void* d_ws, size_t ws_size,
                              hipStream_t stream)
{
  const float* x    = (const float*)d_in[0];
  const float* c    = (const float*)d_in[1];
  const float* Wq   = (const float*)d_in[2];
  const float* bq   = (const float*)d_in[3];
  const float* Wk   = (const float*)d_in[4];
  const float* bk   = (const float*)d_in[5];
  const float* Wv   = (const float*)d_in[6];
  const float* bv   = (const float*)d_in[7];
  const float* Wo   = (const float*)d_in[8];
  const float* bo   = (const float*)d_in[9];
  const float* W1   = (const float*)d_in[10];
  const float* b1   = (const float*)d_in[11];
  const float* W2   = (const float*)d_in[12];
  const float* b2   = (const float*)d_in[13];
  const float* Wada = (const float*)d_in[14];
  const float* bada = (const float*)d_in[15];
  float* out        = (float*)d_out;

  char* w = (char*)d_ws;
  size_t off = 0;
  auto alloc = [&](size_t bytes) {
    size_t o = off; off += (bytes + 255) & ~(size_t)255; return o;
  };
  float*  ada = (float*)(w + alloc((size_t)B_ * ADA_ * 4));
  bf16_t* hb  = (bf16_t*)(w + alloc((size_t)ROWS_ * D_ * 2));
  bf16_t* Wqh = (bf16_t*)(w + alloc((size_t)D_ * D_ * 2));
  bf16_t* Wkh = (bf16_t*)(w + alloc((size_t)D_ * D_ * 2));
  bf16_t* Wvh = (bf16_t*)(w + alloc((size_t)D_ * D_ * 2));
  bf16_t* Woh = (bf16_t*)(w + alloc((size_t)D_ * D_ * 2));
  bf16_t* W1h = (bf16_t*)(w + alloc((size_t)D_ * FF_ * 2));
  bf16_t* W2h = (bf16_t*)(w + alloc((size_t)FF_ * D_ * 2));
  bf16_t* qh  = (bf16_t*)(w + alloc((size_t)ROWS_ * D_ * 2));
  bf16_t* kh  = (bf16_t*)(w + alloc((size_t)ROWS_ * D_ * 2));
  bf16_t* vth = (bf16_t*)(w + alloc((size_t)ROWS_ * D_ * 2));
  bf16_t* om  = (bf16_t*)(w + alloc((size_t)ROWS_ * D_ * 2));
  float*  x1  = (float*)(w + alloc((size_t)ROWS_ * D_ * 4));
  bf16_t* h2  = (bf16_t*)(w + alloc((size_t)ROWS_ * D_ * 2));
  bf16_t* m1  = (bf16_t*)(w + alloc((size_t)ROWS_ * FF_ * 2));
  (void)ws_size; (void)n_in; (void)in_sizes; (void)out_size;

  dim3 blk(256);

  // weight casts
  k_f2bf<<<dim3((D_ * D_ + 255) / 256), blk, 0, stream>>>(Wq, Wqh, D_ * D_);
  k_f2bf<<<dim3((D_ * D_ + 255) / 256), blk, 0, stream>>>(Wk, Wkh, D_ * D_);
  k_f2bf<<<dim3((D_ * D_ + 255) / 256), blk, 0, stream>>>(Wv, Wvh, D_ * D_);
  k_f2bf<<<dim3((D_ * D_ + 255) / 256), blk, 0, stream>>>(Wo, Woh, D_ * D_);
  k_f2bf<<<dim3((D_ * FF_ + 255) / 256), blk, 0, stream>>>(W1, W1h, D_ * FF_);
  k_f2bf<<<dim3((FF_ * D_ + 255) / 256), blk, 0, stream>>>(W2, W2h, FF_ * D_);

  // adaLN params
  k_ada<<<dim3(B_ * ADA_ / 8), blk, 0, stream>>>(c, Wada, bada, ada);

  // LN + modulate (msa) -> hb
  k_lnmod<<<dim3(ROWS_), blk, 0, stream>>>(x, ada, /*shift*/0, /*scale*/D_, hb);

  // QKV projections
  dim3 gD(D_ / TN, ROWS_ / TM);
  k_gemm<0><<<gD, blk, 0, stream>>>(hb, Wqh, bq, ROWS_, D_, D_, nullptr, qh, nullptr, nullptr, 0);
  k_gemm<0><<<gD, blk, 0, stream>>>(hb, Wkh, bk, ROWS_, D_, D_, nullptr, kh, nullptr, nullptr, 0);
  k_gemm<1><<<gD, blk, 0, stream>>>(hb, Wvh, bv, ROWS_, D_, D_, nullptr, vth, nullptr, nullptr, 0);

  // attention (4096 wave-tasks / 8 waves per block)
  k_attn<<<dim3(512), blk, 0, stream>>>(qh, kh, vth, om);

  // x1 = x + gate_msa * (om @ Wo + bo)
  k_gemm<2><<<gD, blk, 0, stream>>>(om, Woh, bo, ROWS_, D_, D_, x1, nullptr, x, ada, 2 * D_);

  // LN + modulate (mlp) -> h2
  k_lnmod<<<dim3(ROWS_), blk, 0, stream>>>(x1, ada, /*shift*/3 * D_, /*scale*/4 * D_, h2);

  // m1 = gelu(h2 @ W1 + b1)
  dim3 gF(FF_ / TN, ROWS_ / TM);
  k_gemm<3><<<gF, blk, 0, stream>>>(h2, W1h, b1, ROWS_, FF_, D_, nullptr, m1, nullptr, nullptr, 0);

  // out = x1 + gate_mlp * (m1 @ W2 + b2)
  k_gemm<4><<<gD, blk, 0, stream>>>(m1, W2h, b2, ROWS_, D_, FF_, out, nullptr, x1, ada, 5 * D_);
}
